// BiBoMoELayer_89996744720725
// MI455X (gfx1250) — matrix operand order repaired and checked by reading, not compiled, not run on hardware
//
#include <hip/hip_runtime.h>

// ---------------- config (matches reference) ----------------
constexpr int Hd  = 1024;   // hidden
constexpr int En  = 8;      // experts total
constexpr int EM  = 4;      // MLP experts
constexpr int Id  = 512;    // moe intermediate
constexpr int KCc = 4;      // conv kernel
constexpr int Tt  = 16384;  // tokens = B*S
constexpr int Ss  = 4096;   // sequence length (power of 2)

constexpr int MT  = 64;        // token rows per expert block (4 WMMA row-tiles)
constexpr int XSS = Hd + 8;    // padded LDS row stride (bf16 elems) for x tile
constexpr int HSS = Id + 8;    // padded LDS row stride for hidden tile

typedef __bf16 bf16;
typedef __attribute__((ext_vector_type(16))) bf16  v16bf;
typedef __attribute__((ext_vector_type(8)))  float v8f;

union ABfrag {
  v16bf v;
  unsigned short u[16];
  uint4 q[2];
};

__device__ __forceinline__ unsigned short f32_to_bf16(float f) {
  unsigned int u = __float_as_uint(f);
  u += 0x7FFFu + ((u >> 16) & 1u);   // round-to-nearest-even
  return (unsigned short)(u >> 16);
}

// ---------------------------------------------------------------------------
// Zero the 8 list counters (slot-major: cnt[slot*4 + expert])
// ---------------------------------------------------------------------------
__global__ void zero_cnt_kernel(int* cnt) {
  if (threadIdx.x < 8) cnt[threadIdx.x] = 0;
}

// ---------------------------------------------------------------------------
// Pack fp32 weights [EM, Kd, Nd] into bf16 WMMA B-fragment layout:
//   dst[(((e*(Nd/16)+nt)*(Kd/32)+kk)*512) + lane*16 + t]
//     = src[e, kk*32 + (t<8 ? kbase + t : 16 + kbase + t - 8), nt*16 + (lane&15)],  kbase = (lane<16 ? 0 : 8)
// so each lane of a wave reads its 16 bf16 B elements as one contiguous 32B load.
// ---------------------------------------------------------------------------
__global__ void pack_kernel(const float* __restrict__ src,
                            unsigned short* __restrict__ dst,
                            int Kd, int Nd) {
  long idx = (long)blockIdx.x * 256 + threadIdx.x;
  long total = (long)EM * Kd * Nd;
  if (idx >= total) return;
  int t    = (int)(idx & 15);
  int lane = (int)((idx >> 4) & 31);
  long chunk = idx >> 9;
  int nkt = Kd / 32;
  int kk = (int)(chunk % nkt);
  long c2 = chunk / nkt;
  int nnt = Nd / 16;
  int nt = (int)(c2 % nnt);
  int e  = (int)(c2 / nnt);
  int kbase = (lane < 16) ? 0 : 8;
  int K = kk * 32 + ((t < 8) ? kbase + t : 16 + kbase + (t - 8));
  int N = nt * 16 + (lane & 15);
  dst[idx] = f32_to_bf16(src[((long)e * Kd + K) * Nd + N]);
}

// ---------------------------------------------------------------------------
// Router + element-wise experts + depthwise causal conv. One block per token.
// ---------------------------------------------------------------------------
__global__ void router_kernel(const float* __restrict__ x,
                              const float* __restrict__ Wr,
                              const float* __restrict__ rbias,
                              const float* __restrict__ conv_w,
                              const float* __restrict__ conv_b,
                              float* __restrict__ out,
                              int* __restrict__ cnt,
                              int* __restrict__ tok,
                              float* __restrict__ wt) {
  __shared__ float xrow[Hd];
  __shared__ float sc[En];
  __shared__ float cws[En];

  int t = blockIdx.x;
  int tid = threadIdx.x;

  // stage x row (1024 floats) into LDS
  {
    const float4* xr4 = (const float4*)(x + (size_t)t * Hd);
    ((float4*)xrow)[tid] = xr4[tid];
  }
  __syncthreads();

  int wave = tid >> 5, lane = tid & 31;
  if (wave < En) {
    float acc = 0.f;
    for (int h = lane; h < Hd; h += 32) acc += xrow[h] * Wr[h * En + wave];
    #pragma unroll
    for (int off = 16; off; off >>= 1) acc += __shfl_xor(acc, off, 32);
    if (lane == 0) sc[wave] = __builtin_amdgcn_rcpf(1.f + __expf(-acc));
  }
  __syncthreads();

  if (tid == 0) {
    int i0 = 0, i1 = 0;
    float v0 = -1e30f, v1 = -1e30f;
    float s[En];
    #pragma unroll
    for (int e = 0; e < En; ++e) {
      s[e] = sc[e];
      float be = s[e] + rbias[e];
      if (be > v0) { v1 = v0; i1 = i0; v0 = be; i0 = e; }
      else if (be > v1) { v1 = be; i1 = e; }
    }
    float w0 = s[i0], w1 = s[i1];
    float inv = 1.f / (w0 + w1 + 1e-9f);
    w0 *= inv; w1 *= inv;
    #pragma unroll
    for (int e = 0; e < En; ++e) cws[e] = 0.f;
    cws[i0] = w0;
    cws[i1] = w1;
    // slot-partitioned compacted lists (slot0 and slot1 launches are disjoint)
    if (i0 < EM) { int p = atomicAdd(&cnt[0 * 4 + i0], 1); tok[(0 * 4 + i0) * Tt + p] = t; wt[(0 * 4 + i0) * Tt + p] = w0; }
    if (i1 < EM) { int p = atomicAdd(&cnt[1 * 4 + i1], 1); tok[(1 * 4 + i1) * Tt + p] = t; wt[(1 * 4 + i1) * Tt + p] = w1; }
  }
  __syncthreads();

  float cid = cws[4] + cws[6];   // identity + noise(identity at eval)
  float crl = cws[7];            // relu expert
  int s_pos = t & (Ss - 1);

  #pragma unroll
  for (int j = 0; j < 4; ++j) {
    int h = tid + j * 256;
    float xv = xrow[h];
    float acc = cid * xv + crl * fmaxf(xv, 0.f) + conv_b[h];
    #pragma unroll
    for (int k = 0; k < KCc; ++k) {
      int d = (KCc - 1) - k;            // conv_w[k] multiplies x[s - (KC-1) + k]
      if (s_pos - d >= 0) {
        float xn = (d == 0) ? xv : x[(size_t)(t - d) * Hd + h];
        acc += conv_w[k * Hd + h] * xn;
      }
    }
    out[(size_t)t * Hd + h] = acc;
  }
}

// ---------------------------------------------------------------------------
// Sparse SwiGLU expert GEMMs with v_wmma_f32_16x16x32_bf16.
// One block = 64-token tile (4 row-tiles) of expert e's list for a given slot.
// Each B fragment is reused by 4 WMMAs (one per row-tile) -> 64 flop/byte from L2.
// Dynamic LDS (~199 KB): x tile (bf16) + hidden tile (bf16), padded rows to
// dodge bank conflicts on the A-fragment gathers.
// ---------------------------------------------------------------------------
__global__ void __launch_bounds__(256)
expert_kernel(const float* __restrict__ x,
              const unsigned short* __restrict__ WgP,
              const unsigned short* __restrict__ WuP,
              const unsigned short* __restrict__ WdP,
              const int* __restrict__ cnt,
              const int* __restrict__ tok,
              const float* __restrict__ wt,
              float* __restrict__ out,
              int slot) {
  int e = blockIdx.y;
  int tile = blockIdx.x;
  int n = cnt[slot * 4 + e];
  if (tile * MT >= n) return;

  extern __shared__ unsigned char smem[];
  unsigned short* xs = (unsigned short*)smem;            // [MT][XSS] bf16
  unsigned short* hs = xs + (size_t)MT * XSS;            // [MT][HSS] bf16
  int*   tks = (int*)(hs + (size_t)MT * HSS);            // [MT]
  float* wts = (float*)(tks + MT);                       // [MT]

  const int* mytok = tok + (slot * 4 + e) * Tt;
  const float* mywt = wt + (slot * 4 + e) * Tt;

  int tid = threadIdx.x;
  if (tid < MT) {
    int r = tile * MT + tid;
    bool valid = r < n;
    tks[tid] = valid ? mytok[r] : mytok[tile * MT];  // clamp pad rows to a real token
    wts[tid] = valid ? mywt[r] : 0.f;
  }
  __syncthreads();

  // gather x rows -> bf16 LDS (coalesced within rows)
  for (int i = tid; i < MT * Hd; i += 256) {
    int r = i >> 10;
    int c = i & (Hd - 1);
    xs[r * XSS + c] = f32_to_bf16(x[(size_t)tks[r] * Hd + c]);
  }
  __syncthreads();

  int wave = tid >> 5, lane = tid & 31;
  int mrow  = lane & 15;              // A row within a 16-row tile
  int kb    = (lane < 16) ? 0 : 8;    // A K-sub-base per ISA 16-bit A layout
  int mrow0 = (lane >> 4) * 8;        // C/D row base
  int ncol  = lane & 15;              // C/D column

  // ---------------- gate + up:  [64,1024] x [1024,512] twice ----------------
  constexpr int NKT1 = Hd / 32;       // 32 K-tiles
  for (int jn = 0; jn < 4; ++jn) {    // 8 waves x 4 = 32 n-tiles over I
    int nt = wave + jn * 8;
    v8f accg[4], accu[4];
    #pragma unroll
    for (int mt = 0; mt < 4; ++mt) { accg[mt] = 0.f; accu[mt] = 0.f; }

    size_t bbase = ((size_t)(e * (Id / 16) + nt)) * NKT1 * 512 + lane * 16;
    for (int kk = 0; kk < NKT1; ++kk) {
      ABfrag bg, bu;
      bg.v = *(const v16bf*)(WgP + bbase + (size_t)kk * 512);
      bu.v = *(const v16bf*)(WuP + bbase + (size_t)kk * 512);
      #pragma unroll
      for (int mt = 0; mt < 4; ++mt) {
        ABfrag a;
        const unsigned short* row = xs + (mt * 16 + mrow) * XSS + kk * 32;
        a.q[0] = *(const uint4*)(row + kb);        // K kb..kb+7
        a.q[1] = *(const uint4*)(row + 16 + kb);   // K 16+kb..16+kb+7
        accg[mt] = __builtin_amdgcn_wmma_f32_16x16x32_bf16(false, a.v, false, bg.v,
                                                           (short)0, accg[mt], false, false);
        accu[mt] = __builtin_amdgcn_wmma_f32_16x16x32_bf16(false, a.v, false, bu.v,
                                                           (short)0, accu[mt], false, false);
      }
    }

    // SwiGLU -> hs (bf16); fast sigmoid via v_rcp_f32
    #pragma unroll
    for (int mt = 0; mt < 4; ++mt) {
      #pragma unroll
      for (int r = 0; r < 8; ++r) {
        float g = accg[mt][r], u = accu[mt][r];
        float sg = __builtin_amdgcn_rcpf(1.f + __expf(-g));
        hs[(mt * 16 + mrow0 + r) * HSS + nt * 16 + ncol] = f32_to_bf16(g * sg * u);
      }
    }
  }
  __syncthreads();

  // ---------------- down: [64,512] x [512,1024] ----------------
  constexpr int NKT2 = Id / 32;       // 16 K-tiles
  for (int jn = 0; jn < 8; ++jn) {    // 8 waves x 8 = 64 n-tiles over H
    int nt = wave + jn * 8;
    v8f accd[4];
    #pragma unroll
    for (int mt = 0; mt < 4; ++mt) accd[mt] = 0.f;

    size_t bbase = ((size_t)(e * (Hd / 16) + nt)) * NKT2 * 512 + lane * 16;
    for (int kk = 0; kk < NKT2; ++kk) {
      ABfrag bd;
      bd.v = *(const v16bf*)(WdP + bbase + (size_t)kk * 512);
      #pragma unroll
      for (int mt = 0; mt < 4; ++mt) {
        ABfrag a;
        const unsigned short* row = hs + (mt * 16 + mrow) * HSS + kk * 32;
        a.q[0] = *(const uint4*)(row + kb);
        a.q[1] = *(const uint4*)(row + 16 + kb);
        accd[mt] = __builtin_amdgcn_wmma_f32_16x16x32_bf16(false, a.v, false, bd.v,
                                                           (short)0, accd[mt], false, false);
      }
    }

    // scaled scatter-add; rows disjoint within a launch -> plain RMW is race-free
    #pragma unroll
    for (int mt = 0; mt < 4; ++mt) {
      #pragma unroll
      for (int r = 0; r < 8; ++r) {
        int mr = mt * 16 + mrow0 + r;
        if (tile * MT + mr < n) {
          out[(size_t)tks[mr] * Hd + nt * 16 + ncol] += wts[mr] * accd[mt][r];
        }
      }
    }
  }
}

// ---------------------------------------------------------------------------
extern "C" void kernel_launch(void* const* d_in, const int* in_sizes, int n_in,
                              void* d_out, int out_size, void* d_ws, size_t ws_size,
                              hipStream_t stream) {
  const float* x      = (const float*)d_in[0];
  const float* Wr     = (const float*)d_in[1];
  const float* rbias  = (const float*)d_in[2];
  const float* Wg     = (const float*)d_in[3];
  const float* Wu     = (const float*)d_in[4];
  const float* Wd     = (const float*)d_in[5];
  const float* conv_w = (const float*)d_in[6];
  const float* conv_b = (const float*)d_in[7];
  float* out = (float*)d_out;

  // workspace layout
  char* ws = (char*)d_ws;
  int*   cnt = (int*)ws;                                   // 8 ints (pad 256)
  int*   tok = (int*)(ws + 256);                           // [2][4][T]
  float* wtl = (float*)(ws + 256 + (size_t)8 * Tt * 4);    // [2][4][T]
  unsigned short* WgP = (unsigned short*)(ws + 256 + (size_t)16 * Tt * 4);
  unsigned short* WuP = WgP + (size_t)EM * Hd * Id;
  unsigned short* WdP = WuP + (size_t)EM * Hd * Id;

  zero_cnt_kernel<<<1, 32, 0, stream>>>(cnt);

  long np = (long)EM * Hd * Id;
  int pb = (int)((np + 255) / 256);
  pack_kernel<<<pb, 256, 0, stream>>>(Wg, WgP, Hd, Id);
  pack_kernel<<<pb, 256, 0, stream>>>(Wu, WuP, Hd, Id);
  pack_kernel<<<pb, 256, 0, stream>>>(Wd, WdP, Id, Hd);

  router_kernel<<<Tt, 256, 0, stream>>>(x, Wr, rbias, conv_w, conv_b, out, cnt, tok, wtl);

  // dynamic LDS: x tile + hidden tile + token/weight staging (~199 KB, <320 KB/WGP)
  size_t smem = ((size_t)MT * XSS + (size_t)MT * HSS) * sizeof(unsigned short)
              + (size_t)MT * (sizeof(int) + sizeof(float));
  dim3 g(Tt / MT, EM);
  expert_kernel<<<g, 256, smem, stream>>>(x, WgP, WuP, WdP, cnt, tok, wtl, out, 0);
  expert_kernel<<<g, 256, smem, stream>>>(x, WgP, WuP, WdP, cnt, tok, wtl, out, 1);
}
